// AttentionBasedCNN_63093069578261
// MI455X (gfx1250) — matrix operand order repaired and checked by reading, not compiled, not run on hardware
//
#include <hip/hip_runtime.h>
#include <hip/hip_bf16.h>

typedef __attribute__((ext_vector_type(16))) _Float16 v16h;
typedef __attribute__((ext_vector_type(8)))  _Float16 v8h;
typedef __attribute__((ext_vector_type(8)))  float    v8f;

#define NB      131072
#define KTOT    576
#define HNU     256
#define EPSF    1e-5f

// workspace layout (bytes)
#define FEAT_BYTES  ((size_t)NB * KTOT * 2)          // 150,994,944
#define FW1H_OFF    FEAT_BYTES
#define FW2H_OFF    (FW1H_OFF + (size_t)HNU * KTOT * 2)   // +294,912
#define GP_OFF      (FW2H_OFF + (size_t)16 * HNU * 2)     // +8,192

// ---------------------------------------------------------------------------
// Kernel 1: weight prep.
//   blocks 0..575   : fw1 f32 -> f16 with K permutation k' = br*288 + i*32 + oc
//   blocks 576..591 : fw2 padded [16][256] f16 tile (rows >=2 are zero)
//   block 592       : Gp[7][7] = wqp^T wkp ; gp[49] = Sqk = wq . wk
// ---------------------------------------------------------------------------
__global__ __launch_bounds__(256) void prep_kernel(
    const float* __restrict__ fw1, const float* __restrict__ fw2,
    const float* __restrict__ wq,  const float* __restrict__ wk,
    const float* __restrict__ wqp, const float* __restrict__ wkp,
    _Float16* __restrict__ fw1h, _Float16* __restrict__ fw2h,
    float* __restrict__ gp)
{
    int bid = blockIdx.x, tid = threadIdx.x;
    if (bid < 576) {
        int g  = bid * 256 + tid;          // < 147456
        int j  = g / KTOT;
        int kp = g - j * KTOT;
        int br = kp / 288;
        int r  = kp - br * 288;
        int i  = r >> 5;                   // 0..8
        int oc = r & 31;                   // 0..31
        int korig = br * 288 + oc * 9 + i;
        fw1h[g] = (_Float16)fw1[j * KTOT + korig];
    } else if (bid < 592) {
        int g = (bid - 576) * 256 + tid;   // < 4096
        int j = g >> 8, k = g & 255;
        fw2h[g] = (j < 2) ? (_Float16)fw2[j * HNU + k] : (_Float16)0.0f;
    } else {
        if (tid < 49) {
            int a = tid / 7, b = tid - a * 7;
            float acc = 0.f;
            #pragma unroll
            for (int i = 0; i < 32; ++i)
                acc = fmaf(wqp[i * 7 + a], wkp[i * 7 + b], acc);
            gp[tid] = acc;
        } else if (tid == 63) {
            float acc = 0.f;
            #pragma unroll
            for (int i = 0; i < 32; ++i) acc = fmaf(wq[i], wk[i], acc);
            gp[49] = acc;
        }
    }
}

// ---------------------------------------------------------------------------
// Kernel 2: both attention branches, one thread per sample.
// Folded math:
//   image:  energy[n][m] = Sqk * x[n] * x[m];  v[n] = wv00 * x[n]
//   param:  energy[n][m] = x_n^T Gp x_m
// BN folded into (cw, cb) per conv. Features written f16, layout
//   feat[s][br*288 + i*32 + oc]  (contiguous 64B per spatial position)
// ---------------------------------------------------------------------------
__global__ __launch_bounds__(256) void branch_kernel(
    const float* __restrict__ x,    const float* __restrict__ xp,
    const float* __restrict__ wv,
    const float* __restrict__ w1,   const float* __restrict__ b1,
    const float* __restrict__ g1,   const float* __restrict__ be1,
    const float* __restrict__ m1,   const float* __restrict__ v1,
    const float* __restrict__ w2,   const float* __restrict__ b2,
    const float* __restrict__ g2,   const float* __restrict__ be2,
    const float* __restrict__ m2,   const float* __restrict__ v2,
    const float* __restrict__ wvp,
    const float* __restrict__ w1p,  const float* __restrict__ b1p,
    const float* __restrict__ g1p,  const float* __restrict__ be1p,
    const float* __restrict__ m1p,  const float* __restrict__ v1p,
    const float* __restrict__ w2p,  const float* __restrict__ b2p,
    const float* __restrict__ g2p,  const float* __restrict__ be2p,
    const float* __restrict__ m2p,  const float* __restrict__ v2p,
    const float* __restrict__ gp,
    _Float16* __restrict__ feat)
{
    __shared__ float s_cw1[32],  s_cb1[32],  s_cw2[1024],  s_cb2[32];
    __shared__ float s_cw1p[224], s_cb1p[32], s_cw2p[1024], s_cb2p[32];
    __shared__ float s_gp[49], s_wvp[49], s_misc[2];
    __shared__ float s_sc2[32], s_sc1p[32], s_sc2p[32];

    int tid = threadIdx.x;
    if (tid < 32) {
        float sc1 = g1[tid] * rsqrtf(v1[tid] + EPSF);
        s_cw1[tid] = sc1 * w1[tid];
        s_cb1[tid] = sc1 * b1[tid] + be1[tid] - m1[tid] * sc1;
        float sc2 = g2[tid] * rsqrtf(v2[tid] + EPSF);
        s_sc2[tid] = sc2;
        s_cb2[tid] = sc2 * b2[tid] + be2[tid] - m2[tid] * sc2;
        float sc1p = g1p[tid] * rsqrtf(v1p[tid] + EPSF);
        s_sc1p[tid] = sc1p;
        s_cb1p[tid] = sc1p * b1p[tid] + be1p[tid] - m1p[tid] * sc1p;
        float sc2p = g2p[tid] * rsqrtf(v2p[tid] + EPSF);
        s_sc2p[tid] = sc2p;
        s_cb2p[tid] = sc2p * b2p[tid] + be2p[tid] - m2p[tid] * sc2p;
    } else if (tid >= 64 && tid < 113) {
        s_gp[tid - 64] = gp[tid - 64];
    } else if (tid >= 128 && tid < 177) {
        s_wvp[tid - 128] = wvp[tid - 128];
    } else if (tid == 200) {
        s_misc[0] = gp[49];     // Sqk
        s_misc[1] = wv[0];      // wv00
    }
    __syncthreads();
    for (int idx = tid; idx < 1024; idx += 256) {
        s_cw2[idx]  = s_sc2[idx >> 5]  * w2[idx];
        s_cw2p[idx] = s_sc2p[idx >> 5] * w2p[idx];
    }
    for (int idx = tid; idx < 224; idx += 256)
        s_cw1p[idx] = s_sc1p[idx / 7] * w1p[idx];
    __syncthreads();

    int s = blockIdx.x * 256 + tid;
    _Float16* frow = feat + (size_t)s * KTOT;
    float Sqk = s_misc[0], wv00 = s_misc[1];

    // ---------------- image branch (CIN = 1) ----------------
    {
        const float* xr = x + (size_t)s * 9;
        float xi[9];
        #pragma unroll
        for (int n = 0; n < 9; ++n) xi[n] = xr[n];

        for (int i = 0; i < 9; ++i) {
            float f  = Sqk * xi[i];
            float mx = f * xi[0];
            #pragma unroll
            for (int m = 1; m < 9; ++m) mx = fmaxf(mx, f * xi[m]);
            float den = 0.f, num = 0.f;
            #pragma unroll
            for (int m = 0; m < 9; ++m) {
                float ex = __expf(f * xi[m] - mx);
                den += ex;
                num = fmaf(ex, xi[m], num);
            }
            float oi = wv00 * num / den;
            float r1[32];
            #pragma unroll
            for (int oc = 0; oc < 32; ++oc)
                r1[oc] = fmaxf(fmaf(s_cw1[oc], oi, s_cb1[oc]), 0.f);
            alignas(16) _Float16 hb[32];
            #pragma unroll
            for (int oc2 = 0; oc2 < 32; ++oc2) {
                float acc = s_cb2[oc2];
                #pragma unroll
                for (int oc = 0; oc < 32; ++oc)
                    acc = fmaf(s_cw2[oc2 * 32 + oc], r1[oc], acc);
                hb[oc2] = (_Float16)fmaxf(acc, 0.f);
            }
            v8h* dst = (v8h*)(frow + i * 32);
            #pragma unroll
            for (int q = 0; q < 4; ++q) dst[q] = ((const v8h*)hb)[q];
        }
    }

    // ---------------- param branch (CP = 7) ----------------
    {
        const float* xpr = xp + (size_t)s * 63;
        float xpl[63];
        #pragma unroll
        for (int t = 0; t < 63; ++t) xpl[t] = xpr[t];

        for (int i = 0; i < 9; ++i) {
            float t7[7];
            #pragma unroll
            for (int b = 0; b < 7; ++b) {
                float acc = 0.f;
                #pragma unroll
                for (int a = 0; a < 7; ++a)
                    acc = fmaf(xpl[a * 9 + i], s_gp[a * 7 + b], acc);
                t7[b] = acc;
            }
            float er[9];
            #pragma unroll
            for (int m = 0; m < 9; ++m) {
                float acc = 0.f;
                #pragma unroll
                for (int b = 0; b < 7; ++b)
                    acc = fmaf(t7[b], xpl[b * 9 + m], acc);
                er[m] = acc;
            }
            float mx = er[0];
            #pragma unroll
            for (int m = 1; m < 9; ++m) mx = fmaxf(mx, er[m]);
            float den = 0.f;
            float o7[7] = {0.f, 0.f, 0.f, 0.f, 0.f, 0.f, 0.f};
            #pragma unroll
            for (int m = 0; m < 9; ++m) {
                float exm = __expf(er[m] - mx);
                den += exm;
                #pragma unroll
                for (int c = 0; c < 7; ++c) {
                    float vcm = 0.f;
                    #pragma unroll
                    for (int a = 0; a < 7; ++a)
                        vcm = fmaf(s_wvp[c * 7 + a], xpl[a * 9 + m], vcm);
                    o7[c] = fmaf(vcm, exm, o7[c]);
                }
            }
            float inv = 1.f / den;
            #pragma unroll
            for (int c = 0; c < 7; ++c) o7[c] *= inv;
            float r1[32];
            #pragma unroll
            for (int oc = 0; oc < 32; ++oc) {
                float acc = s_cb1p[oc];
                #pragma unroll
                for (int c = 0; c < 7; ++c)
                    acc = fmaf(s_cw1p[oc * 7 + c], o7[c], acc);
                r1[oc] = fmaxf(acc, 0.f);
            }
            alignas(16) _Float16 hb[32];
            #pragma unroll
            for (int oc2 = 0; oc2 < 32; ++oc2) {
                float acc = s_cb2p[oc2];
                #pragma unroll
                for (int oc = 0; oc < 32; ++oc)
                    acc = fmaf(s_cw2p[oc2 * 32 + oc], r1[oc], acc);
                hb[oc2] = (_Float16)fmaxf(acc, 0.f);
            }
            v8h* dst = (v8h*)(frow + 288 + i * 32);
            #pragma unroll
            for (int q = 0; q < 4; ++q) dst[q] = ((const v8h*)hb)[q];
        }
    }
}

// ---------------------------------------------------------------------------
// Kernel 3: fused FC1 (WMMA f16->f32) + ReLU + FC2 (WMMA vs padded fw2 tile)
//           + bias + sigmoid. 512 threads = 16 waves; block handles 16 samples.
// ---------------------------------------------------------------------------
__global__ __launch_bounds__(512) void fc_head_kernel(
    const _Float16* __restrict__ feat, const _Float16* __restrict__ fw1h,
    const _Float16* __restrict__ fw2h, const float* __restrict__ fb1,
    const float* __restrict__ fb2, float* __restrict__ out)
{
    __shared__ _Float16 ash[16 * HNU];   // h activations, f16, [row][hidden]

    int tid  = threadIdx.x;
    int wave = tid >> 5;
    int lane = tid & 31;
    int half = lane >> 4;                // K-half selector
    int lid  = lane & 15;
    int m0   = blockIdx.x << 4;          // sample tile base
    int n0   = wave << 4;                // hidden tile base

    const _Float16* arow = feat + (size_t)(m0 + lid) * KTOT + half * 16;
    const _Float16* brow = fw1h + (size_t)(n0 + lid) * KTOT + half * 16;

    v8f c = {};
    #pragma unroll
    for (int ko = 0; ko < 18; ++ko) {
        v8h lo = *(const v8h*)(arow + ko * 32);
        v8h hi = *(const v8h*)(arow + ko * 32 + 8);
        v16h a;
        #pragma unroll
        for (int e = 0; e < 8; ++e) { a[e] = lo[e]; a[e + 8] = hi[e]; }
        v16h b = *(const v16h*)(brow + ko * 32);
        c = __builtin_amdgcn_wmma_f32_16x16x32_f16(
                false, a, false, b, (short)0, c, false, false);
    }

    // bias + ReLU, stage h in LDS (f16)
    float bias = fb1[n0 + lid];
    #pragma unroll
    for (int r = 0; r < 8; ++r) {
        float h = fmaxf(c[r] + bias, 0.f);
        ash[(half * 8 + r) * HNU + n0 + lid] = (_Float16)h;
    }
    __syncthreads();

    if (wave == 0) {
        const _Float16* a2  = ash  + lid * HNU + half * 16;
        const _Float16* b2r = fw2h + lid * HNU + half * 16;
        v8f c2 = {};
        #pragma unroll
        for (int ko = 0; ko < 8; ++ko) {
            v8h lo = *(const v8h*)(a2 + ko * 32);
            v8h hi = *(const v8h*)(a2 + ko * 32 + 8);
            v16h a;
            #pragma unroll
            for (int e = 0; e < 8; ++e) { a[e] = lo[e]; a[e + 8] = hi[e]; }
            v16h b = *(const v16h*)(b2r + ko * 32);
            c2 = __builtin_amdgcn_wmma_f32_16x16x32_f16(
                    false, a, false, b, (short)0, c2, false, false);
        }
        if (lid < 2) {
            float bb = fb2[lid];
            #pragma unroll
            for (int r = 0; r < 8; ++r) {
                float v = c2[r] + bb;
                out[(size_t)(m0 + half * 8 + r) * 2 + lid] =
                    1.f / (1.f + __expf(-v));
            }
        }
    }
}

// ---------------------------------------------------------------------------
extern "C" void kernel_launch(void* const* d_in, const int* in_sizes, int n_in,
                              void* d_out, int out_size, void* d_ws, size_t ws_size,
                              hipStream_t stream) {
    const float* x    = (const float*)d_in[0];
    const float* xp   = (const float*)d_in[1];
    const float* wq   = (const float*)d_in[2];
    const float* wk   = (const float*)d_in[3];
    const float* wv   = (const float*)d_in[4];
    const float* w1   = (const float*)d_in[5];
    const float* b1   = (const float*)d_in[6];
    const float* g1   = (const float*)d_in[7];
    const float* be1  = (const float*)d_in[8];
    const float* m1   = (const float*)d_in[9];
    const float* v1   = (const float*)d_in[10];
    const float* w2   = (const float*)d_in[11];
    const float* b2   = (const float*)d_in[12];
    const float* g2   = (const float*)d_in[13];
    const float* be2  = (const float*)d_in[14];
    const float* m2   = (const float*)d_in[15];
    const float* v2   = (const float*)d_in[16];
    const float* wqp  = (const float*)d_in[17];
    const float* wkp  = (const float*)d_in[18];
    const float* wvp  = (const float*)d_in[19];
    const float* w1p  = (const float*)d_in[20];
    const float* b1p  = (const float*)d_in[21];
    const float* g1p  = (const float*)d_in[22];
    const float* be1p = (const float*)d_in[23];
    const float* m1p  = (const float*)d_in[24];
    const float* v1p  = (const float*)d_in[25];
    const float* w2p  = (const float*)d_in[26];
    const float* b2p  = (const float*)d_in[27];
    const float* g2p  = (const float*)d_in[28];
    const float* be2p = (const float*)d_in[29];
    const float* m2p  = (const float*)d_in[30];
    const float* v2p  = (const float*)d_in[31];
    const float* fw1  = (const float*)d_in[32];
    const float* fb1  = (const float*)d_in[33];
    const float* fw2  = (const float*)d_in[34];
    const float* fb2  = (const float*)d_in[35];

    char* ws = (char*)d_ws;
    _Float16* feat = (_Float16*)(ws);
    _Float16* fw1h = (_Float16*)(ws + FW1H_OFF);
    _Float16* fw2h = (_Float16*)(ws + FW2H_OFF);
    float*    gp   = (float*)   (ws + GP_OFF);

    prep_kernel<<<593, 256, 0, stream>>>(fw1, fw2, wq, wk, wqp, wkp,
                                         fw1h, fw2h, gp);

    branch_kernel<<<NB / 256, 256, 0, stream>>>(
        x, xp, wv,
        w1, b1, g1, be1, m1, v1, w2, b2, g2, be2, m2, v2,
        wvp, w1p, b1p, g1p, be1p, m1p, v1p, w2p, b2p, g2p, be2p, m2p, v2p,
        gp, feat);

    fc_head_kernel<<<NB / 16, 512, 0, stream>>>(
        feat, fw1h, fw2h, fb1, fb2, (float*)d_out);
}
